// BilinearDiscriminator_32392643347090
// MI455X (gfx1250) — matrix operand order repaired and checked by reading, not compiled, not run on hardware
//
#include <hip/hip_runtime.h>
#include <hip/hip_bf16.h>

// ---------------------------------------------------------------------------
// BilinearDiscriminator: out = sigmoid( (x*mx) @ W^T @ (y*my)^T )
// Refactored: z = (y*my) @ W   [M,D]  (tiny GEMM)
//             out = sigmoid( (x*mx) @ z^T )  [N,M]  (big GEMM, WMMA bf16)
// ---------------------------------------------------------------------------

typedef __attribute__((ext_vector_type(16))) __bf16 v16bf;
typedef __attribute__((ext_vector_type(8)))  __bf16 v8bf;
typedef __attribute__((ext_vector_type(4)))  __bf16 v4bf;
typedef __attribute__((ext_vector_type(8)))  float  v8f;

constexpr int kN = 8192;
constexpr int kM = 8192;
constexpr int kD = 512;

// ---------------- prep: fused dropout + f32 -> bf16 ------------------------
__global__ __launch_bounds__(256) void dropout_to_bf16(const float* __restrict__ x,
                                                       const float* __restrict__ m,
                                                       __bf16* __restrict__ o, int n4) {
  int i = blockIdx.x * 256 + threadIdx.x;
  if (i >= n4) return;
  const float4 xv = reinterpret_cast<const float4*>(x)[i];
  const float4 mv = reinterpret_cast<const float4*>(m)[i];
  v4bf r;
  r[0] = (__bf16)(xv.x * mv.x);
  r[1] = (__bf16)(xv.y * mv.y);
  r[2] = (__bf16)(xv.z * mv.z);
  r[3] = (__bf16)(xv.w * mv.w);
  reinterpret_cast<v4bf*>(o)[i] = r;
}

// ---------------- prep: W [D,D] f32 -> Wt [D,D] bf16 transposed ------------
// Wt[k*D + j] = (bf16) W[j*D + k], so GEMM1's B operand is row-major [Ncol,K].
__global__ __launch_bounds__(256) void w_transpose_bf16(const float* __restrict__ W,
                                                        __bf16* __restrict__ Wt) {
  __shared__ __bf16 tile[32][33];
  const int tx = threadIdx.x & 31;
  const int ty = threadIdx.x >> 5;            // 8 rows per pass
  const int bj = blockIdx.y * 32;             // source row block (j)
  const int bk = blockIdx.x * 32;             // source col block (k)
  #pragma unroll
  for (int r = ty; r < 32; r += 8)
    tile[r][tx] = (__bf16)W[(bj + r) * kD + bk + tx];
  __syncthreads();
  #pragma unroll
  for (int r = ty; r < 32; r += 8)
    Wt[(bk + r) * kD + bj + tx] = tile[tx][r];
}

// ---------------- WMMA fragment loaders (ISA 7.12.2 layouts) ---------------
// A 16x32 bf16: lane (l&15)=row M, g=l>>4; elems 0..7 -> K=g*8+e,
//               elems 8..15 -> K=16+g*8+(e-8).  ld = kD, rows contiguous.
__device__ inline v16bf load_a_frag(const __bf16* __restrict__ A, int row, int k0, int g) {
  const __bf16* p = A + (size_t)row * kD + k0 + g * 8;
  v8bf lo = *reinterpret_cast<const v8bf*>(p);
  v8bf hi = *reinterpret_cast<const v8bf*>(p + 16);
  return __builtin_shufflevector(lo, hi, 0, 1, 2, 3, 4, 5, 6, 7,
                                         8, 9, 10, 11, 12, 13, 14, 15);
}

// B 32x16 bf16 stored as row-major [Ncol, K]: lane owns column (l&15),
// elems 0..7 -> K = g*8 + e, elems 8..15 -> K = 16 + g*8 + (e-8) (same as A).
__device__ inline v16bf load_b_frag(const __bf16* __restrict__ B, int col, int k0, int g) {
  const __bf16* p = B + (size_t)col * kD + k0 + g * 8;
  v8bf lo = *reinterpret_cast<const v8bf*>(p);
  v8bf hi = *reinterpret_cast<const v8bf*>(p + 16);
  return __builtin_shufflevector(lo, hi, 0, 1, 2, 3, 4, 5, 6, 7,
                                         8, 9, 10, 11, 12, 13, 14, 15);
}

// ---------------- GEMM1: z[m,k] = sum_j yd[m,j] * Wt[k,j] ------------------
// Block: 256 threads = 8 waves (2x4), wave tile 64x64 -> block tile 128x256.
__global__ __launch_bounds__(256) void gemm1_yW(const __bf16* __restrict__ yd,
                                                const __bf16* __restrict__ Wt,
                                                __bf16* __restrict__ z) {
  const int lane = threadIdx.x & 31;
  const int wave = threadIdx.x >> 5;
  const int cl   = lane & 15;
  const int g    = lane >> 4;
  const int mBase = blockIdx.y * 128 + (wave >> 2) * 64;   // rows of z (m)
  const int nBase = blockIdx.x * 256 + (wave & 3) * 64;    // cols of z (k)

  v8f c[4][4] = {};
  for (int k0 = 0; k0 < kD; k0 += 32) {
    v16bf a[4], b[4];
    #pragma unroll
    for (int f = 0; f < 4; ++f) a[f] = load_a_frag(yd, mBase + f * 16 + cl, k0, g);
    #pragma unroll
    for (int f = 0; f < 4; ++f) b[f] = load_b_frag(Wt, nBase + f * 16 + cl, k0, g);
    #pragma unroll
    for (int i = 0; i < 4; ++i)
      #pragma unroll
      for (int j = 0; j < 4; ++j)
        c[i][j] = __builtin_amdgcn_wmma_f32_16x16x32_bf16(
            false, a[i], false, b[j], (short)0, c[i][j], false, false);
  }
  // C/D layout: lane -> N=cl, vgpr r -> M=g*8+r
  #pragma unroll
  for (int i = 0; i < 4; ++i)
    #pragma unroll
    for (int j = 0; j < 4; ++j)
      #pragma unroll
      for (int r = 0; r < 8; ++r)
        z[(size_t)(mBase + i * 16 + g * 8 + r) * kD + nBase + j * 16 + cl] =
            (__bf16)c[i][j][r];
}

// ---------------- GEMM2: out[n,m] = sigmoid( sum_k xd[n,k] * z[m,k] ) ------
__global__ __launch_bounds__(256) void gemm2_sigmoid(const __bf16* __restrict__ xd,
                                                     const __bf16* __restrict__ z,
                                                     float* __restrict__ out) {
  const int lane = threadIdx.x & 31;
  const int wave = threadIdx.x >> 5;
  const int cl   = lane & 15;
  const int g    = lane >> 4;
  const int nBase = blockIdx.y * 128 + (wave >> 2) * 64;   // rows of out (n)
  const int mBase = blockIdx.x * 256 + (wave & 3) * 64;    // cols of out (m)

  v8f c[4][4] = {};
  for (int k0 = 0; k0 < kD; k0 += 32) {
    v16bf a[4], b[4];
    #pragma unroll
    for (int f = 0; f < 4; ++f) a[f] = load_a_frag(xd, nBase + f * 16 + cl, k0, g);
    #pragma unroll
    for (int f = 0; f < 4; ++f) b[f] = load_b_frag(z, mBase + f * 16 + cl, k0, g);
    #pragma unroll
    for (int i = 0; i < 4; ++i)
      #pragma unroll
      for (int j = 0; j < 4; ++j)
        c[i][j] = __builtin_amdgcn_wmma_f32_16x16x32_bf16(
            false, a[i], false, b[j], (short)0, c[i][j], false, false);
  }
  #pragma unroll
  for (int i = 0; i < 4; ++i)
    #pragma unroll
    for (int j = 0; j < 4; ++j)
      #pragma unroll
      for (int r = 0; r < 8; ++r) {
        float v = c[i][j][r];
        float s = 1.0f / (1.0f + __expf(-v));
        out[(size_t)(nBase + i * 16 + g * 8 + r) * kM + mBase + j * 16 + cl] = s;
      }
}

// ---------------------------------------------------------------------------
extern "C" void kernel_launch(void* const* d_in, const int* in_sizes, int n_in,
                              void* d_out, int out_size, void* d_ws, size_t ws_size,
                              hipStream_t stream) {
  const float* x  = (const float*)d_in[0];
  const float* y  = (const float*)d_in[1];
  const float* mx = (const float*)d_in[2];
  const float* my = (const float*)d_in[3];
  const float* W  = (const float*)d_in[4];
  float* out = (float*)d_out;

  char* ws = (char*)d_ws;
  __bf16* xd = (__bf16*)(ws);                                         //  8 MB
  __bf16* yd = (__bf16*)(ws + (size_t)kN * kD * 2);                   //  8 MB
  __bf16* Wt = (__bf16*)(ws + (size_t)(kN + kM) * kD * 2);            // .5 MB
  __bf16* z  = (__bf16*)(ws + (size_t)(kN + kM) * kD * 2
                            + (size_t)kD * kD * 2);                   //  8 MB

  const int n4x = kN * kD / 4;
  const int n4y = kM * kD / 4;
  dropout_to_bf16<<<(n4x + 255) / 256, 256, 0, stream>>>(x, mx, xd, n4x);
  dropout_to_bf16<<<(n4y + 255) / 256, 256, 0, stream>>>(y, my, yd, n4y);
  w_transpose_bf16<<<dim3(kD / 32, kD / 32), 256, 0, stream>>>(W, Wt);

  gemm1_yW<<<dim3(kD / 256, kM / 128), 256, 0, stream>>>(yd, Wt, z);
  gemm2_sigmoid<<<dim3(kM / 256, kN / 128), 256, 0, stream>>>(xd, z, out);
}
